// HPGRelMsgGATLayer_63402307223554
// MI455X (gfx1250) — matrix operand order repaired and checked by reading, not compiled, not run on hardware
//
#include <hip/hip_runtime.h>
#include <hip/hip_bf16.h>
#include <stdint.h>

// ---------------- problem constants ----------------
#define NN 40000
#define EE 640000
#define INF_ 128
#define OUTF 32
#define HH 8
#define NCOLS 272            // 256 msg cols + 8 a_src + 8 a_dst
#define NT 17                // 272/16 column tiles
#define MT 2500              // 40000/16 row tiles

typedef _Float16 v16h __attribute__((ext_vector_type(16)));
typedef float    v8f  __attribute__((ext_vector_type(8)));

union Frag16 { uint4 u[2]; v16h v; };

// ---------------- ws layout (bytes) ----------------
#define OFF_WT    0u                      // 272*128 f16          = 69,632
#define OFF_H16   69632u                  // 40000*128 f16        = 10,240,000
#define OFF_OUTC  10309632u               // 40000*272 f32        = 43,520,000
#define OFF_ATTN  53829632u               // 640000*8 f32         = 20,480,000
#define OFF_SMAX  74309632u               // 40000*8 u32          = 1,280,000
#define OFF_SSUM  75589632u               // 40000*8 f32          = 1,280,000
#define OFF_ACC   76869632u               // 40000*256 f32        = 40,960,000
// total ~117.8 MB

// ---------------- K-: zero init ----------------
__global__ void zero_u32_kernel(uint32_t* p, int count) {
    int t = blockIdx.x * blockDim.x + threadIdx.x;
    if (t < count) p[t] = 0u;
}

// ---------------- K0: h -> f16 ----------------
__global__ void cvt_h_kernel(const float* __restrict__ h, _Float16* __restrict__ h16) {
    int t = blockIdx.x * blockDim.x + threadIdx.x;
    if (t < NN * INF_) h16[t] = (_Float16)h[t];
}

// ---------------- K1: build WcatT (B transposed, [272][128]) ----------------
// rows 0..255  : W_msg[:128] column c           -> WcatT[c][k]   = W_msg[k*256+c]
// rows 256..263: folded attention_src weights   -> Σ_f W_node[k][h*32+f]*att_src[h][f]
// rows 264..271: folded attention_dst weights
__global__ void build_wt_kernel(const float* __restrict__ W_node,
                                const float* __restrict__ att_src,
                                const float* __restrict__ att_dst,
                                const float* __restrict__ W_msg,
                                _Float16* __restrict__ Wt) {
    int t = blockIdx.x * blockDim.x + threadIdx.x;
    if (t >= NCOLS * INF_) return;
    int n = t >> 7;          // row of WcatT (output column)
    int k = t & 127;         // input feature
    float v;
    if (n < 256) {
        v = W_msg[k * 256 + n];
    } else if (n < 264) {
        int hh = n - 256;
        float s = 0.f;
        #pragma unroll 8
        for (int f = 0; f < OUTF; ++f)
            s += W_node[k * 256 + hh * 32 + f] * att_src[hh * 32 + f];
        v = s;
    } else {
        int hh = n - 264;
        float s = 0.f;
        #pragma unroll 8
        for (int f = 0; f < OUTF; ++f)
            s += W_node[k * 256 + hh * 32 + f] * att_dst[hh * 32 + f];
        v = s;
    }
    Wt[n * 128 + k] = (_Float16)v;
}

// ---------------- K2: WMMA GEMM OutCat[40000][272] = h16 @ Wcat ----------------
// Block = 256 threads (8 waves), one 16-row tile per block.
// Static tile assignment (no dynamic-indexed register arrays -> no scratch spills):
//   acc0: col tile `wave`      (0..7)
//   acc1: col tile `wave+8`    (8..15)
//   acc2: col tile 16 — computed redundantly by ALL waves (WMMA stays
//         unconditional, EXEC all-ones), stored only by wave 0.
__global__ __launch_bounds__(256) void gemm_wmma_kernel(const _Float16* __restrict__ h16,
                                                        const _Float16* __restrict__ Wt,
                                                        float* __restrict__ OutCat) {
    __shared__ __align__(16) _Float16 lW[NCOLS * 64];   // one K-half of WcatT (34,816 B)

    const int wave    = threadIdx.x >> 5;
    const int lane    = threadIdx.x & 31;
    const int mt      = blockIdx.x;
    const int halfsel = lane >> 4;                      // 0 or 1
    const int lrow16  = lane & 15;

    const int row = mt * 16 + lrow16;
    const uint4* hrow = reinterpret_cast<const uint4*>(h16 + (size_t)row * 128); // 16 uint4/row

    const int nb0 = (wave)     * 16 + lrow16;           // B row index per tile
    const int nb1 = (wave + 8) * 16 + lrow16;
    const int nb2 = 256 + lrow16;

    v8f acc0 = (v8f){0.f,0.f,0.f,0.f,0.f,0.f,0.f,0.f};
    v8f acc1 = (v8f){0.f,0.f,0.f,0.f,0.f,0.f,0.f,0.f};
    v8f acc2 = (v8f){0.f,0.f,0.f,0.f,0.f,0.f,0.f,0.f};

    const uint4* Wt4 = reinterpret_cast<const uint4*>(Wt);
    uint4* lW4 = reinterpret_cast<uint4*>(lW);

    #pragma unroll
    for (int kh = 0; kh < 2; ++kh) {
        __syncthreads();
        // stage K-half kh of WcatT: halves [kh*64, kh*64+63] of every row
        for (int i = threadIdx.x; i < NCOLS * 8; i += 256) {
            int n = i >> 3, j = i & 7;
            lW4[i] = Wt4[n * 16 + kh * 8 + j];
        }
        __syncthreads();

        #pragma unroll
        for (int ksl = 0; ksl < 2; ++ksl) {             // global K-step = kh*2+ksl
            // A fragment: lanes 0-15 K=kb..kb+15, lanes 16-31 K=kb+16..kb+31
            Frag16 fa;
            int ua = (kh * 2 + ksl) * 4 + (halfsel << 1);
            fa.u[0] = hrow[ua];
            fa.u[1] = hrow[ua + 1];

            // B fragments: lanes 0-15 K=kb..kb+15, lanes 16-31 K=kb+16..kb+31
            const int ub = ksl * 4 + (halfsel << 1);

            Frag16 fb0;
            const uint4* l0 = reinterpret_cast<const uint4*>(lW + nb0 * 64);
            fb0.u[0] = l0[ub]; fb0.u[1] = l0[ub + 1];
            acc0 = __builtin_amdgcn_wmma_f32_16x16x32_f16(
                false, fa.v, false, fb0.v, (short)0, acc0, false, false);

            Frag16 fb1;
            const uint4* l1 = reinterpret_cast<const uint4*>(lW + nb1 * 64);
            fb1.u[0] = l1[ub]; fb1.u[1] = l1[ub + 1];
            acc1 = __builtin_amdgcn_wmma_f32_16x16x32_f16(
                false, fa.v, false, fb1.v, (short)0, acc1, false, false);

            Frag16 fb2;
            const uint4* l2 = reinterpret_cast<const uint4*>(lW + nb2 * 64);
            fb2.u[0] = l2[ub]; fb2.u[1] = l2[ub + 1];
            acc2 = __builtin_amdgcn_wmma_f32_16x16x32_f16(
                false, fa.v, false, fb2.v, (short)0, acc2, false, false);
        }
    }

    // store D: VGPR r -> M = r + (lane<16 ? 0 : 8), N = lane%16
    const int rbase = mt * 16 + (halfsel << 3);
    const int col0  = wave * 16 + lrow16;
    const int col1  = (wave + 8) * 16 + lrow16;
    const int col2  = 256 + lrow16;

    #pragma unroll
    for (int r = 0; r < 8; ++r)
        OutCat[(size_t)(rbase + r) * NCOLS + col0] = acc0[r];
    #pragma unroll
    for (int r = 0; r < 8; ++r)
        OutCat[(size_t)(rbase + r) * NCOLS + col1] = acc1[r];
    if (wave == 0) {
        #pragma unroll
        for (int r = 0; r < 8; ++r)
            OutCat[(size_t)(rbase + r) * NCOLS + col2] = acc2[r];
    }
}

// ---------------- K3: edge attention + leakyrelu + segment max ----------------
__device__ __forceinline__ unsigned f2key(float x) {
    unsigned ux = __float_as_uint(x);
    return (ux & 0x80000000u) ? ~ux : (ux | 0x80000000u);
}
__device__ __forceinline__ float key2f(unsigned key) {
    unsigned ux = (key & 0x80000000u) ? (key & 0x7FFFFFFFu) : ~key;
    return __uint_as_float(ux);
}

__global__ void attn_kernel(const int* __restrict__ src, const int* __restrict__ dst,
                            const float* __restrict__ edge_feat,
                            const float* __restrict__ W_edge,
                            const float* __restrict__ OutCat,
                            float* __restrict__ attnbuf,
                            unsigned* __restrict__ smaxkey) {
    int t = blockIdx.x * blockDim.x + threadIdx.x;
    if (t >= EE * HH) return;
    int e = t >> 3, hh = t & 7;
    int s = src[e], d = dst[e];
    float a = OutCat[(size_t)s * NCOLS + 256 + hh]
            + OutCat[(size_t)d * NCOLS + 264 + hh]
            + edge_feat[e] * W_edge[hh];
    a = (a > 0.f) ? a : 0.2f * a;     // leaky_relu 0.2
    attnbuf[t] = a;
    atomicMax(&smaxkey[d * HH + hh], f2key(a));
}

// ---------------- K4: exp(attn - max) + segment sum ----------------
__global__ void expsum_kernel(const int* __restrict__ dst,
                              const unsigned* __restrict__ smaxkey,
                              float* __restrict__ attnbuf,
                              float* __restrict__ ssum) {
    int t = blockIdx.x * blockDim.x + threadIdx.x;
    if (t >= EE * HH) return;
    int e = t >> 3, hh = t & 7;
    int d = dst[e];
    float m  = key2f(smaxkey[d * HH + hh]);
    float ex = __expf(attnbuf[t] - m);
    attnbuf[t] = ex;
    atomicAdd(&ssum[d * HH + hh], ex);
}

// ---------------- K5: message assembly + normalized scatter-add ----------------
// one block (256 threads) per edge; thread c handles output channel c (h = c>>5)
__global__ __launch_bounds__(256) void scatter_kernel(const int* __restrict__ src,
                                                      const int* __restrict__ dst,
                                                      const float* __restrict__ edge_feat,
                                                      const float* __restrict__ OutCat,
                                                      const float* __restrict__ W_msg,
                                                      const float* __restrict__ alpha,
                                                      const float* __restrict__ ssum,
                                                      float* __restrict__ accum) {
    int e = blockIdx.x;
    __shared__ float coef[HH];
    __shared__ int   ssrc, sdst;
    __shared__ float sef;
    if (threadIdx.x == 0) { ssrc = src[e]; sdst = dst[e]; sef = edge_feat[e]; }
    __syncthreads();
    if (threadIdx.x < HH) {
        float s = ssum[sdst * HH + threadIdx.x];
        coef[threadIdx.x] = alpha[(size_t)e * HH + threadIdx.x] / fmaxf(s, 1e-12f);
    }
    __syncthreads();
    int c = threadIdx.x;      // 0..255
    float val = (OutCat[(size_t)ssrc * NCOLS + c] + sef * W_msg[128 * 256 + c]) * coef[c >> 5];
    atomicAdd(&accum[(size_t)sdst * 256 + c], val);
}

// ---------------- K6: mean over heads ----------------
__global__ void mean_kernel(const float* __restrict__ accum, float* __restrict__ out) {
    int t = blockIdx.x * blockDim.x + threadIdx.x;
    if (t >= NN * OUTF) return;
    int n = t >> 5, f = t & 31;
    float s = 0.f;
    #pragma unroll
    for (int hh = 0; hh < HH; ++hh) s += accum[(size_t)n * 256 + hh * 32 + f];
    out[t] = s * 0.125f;
}

// ---------------- launch ----------------
extern "C" void kernel_launch(void* const* d_in, const int* in_sizes, int n_in,
                              void* d_out, int out_size, void* d_ws, size_t ws_size,
                              hipStream_t stream) {
    (void)in_sizes; (void)n_in; (void)out_size; (void)ws_size;

    const float* h         = (const float*)d_in[0];
    const int*   ei        = (const int*)  d_in[1];
    const float* edge_feat = (const float*)d_in[2];
    const float* W_node    = (const float*)d_in[3];
    const float* W_edge    = (const float*)d_in[4];
    const float* att_src   = (const float*)d_in[5];
    const float* att_dst   = (const float*)d_in[6];
    const float* W_msg     = (const float*)d_in[7];
    float*       out       = (float*)d_out;

    const int* src = ei;
    const int* dst = ei + EE;

    char* ws = (char*)d_ws;
    _Float16* Wt      = (_Float16*)(ws + OFF_WT);
    _Float16* h16     = (_Float16*)(ws + OFF_H16);
    float*    OutCat  = (float*)   (ws + OFF_OUTC);
    float*    attnbuf = (float*)   (ws + OFF_ATTN);
    unsigned* smaxkey = (unsigned*)(ws + OFF_SMAX);
    float*    ssum    = (float*)   (ws + OFF_SSUM);
    float*    accum   = (float*)   (ws + OFF_ACC);

    // zero smaxkey + ssum + accum (contiguous region)
    {
        int cnt = (NN * HH) + (NN * HH) + (NN * 256);
        zero_u32_kernel<<<(cnt + 255) / 256, 256, 0, stream>>>((uint32_t*)smaxkey, cnt);
    }
    cvt_h_kernel<<<(NN * INF_ + 255) / 256, 256, 0, stream>>>(h, h16);
    build_wt_kernel<<<(NCOLS * INF_ + 255) / 256, 256, 0, stream>>>(W_node, att_src, att_dst, W_msg, Wt);
    gemm_wmma_kernel<<<MT, 256, 0, stream>>>(h16, Wt, OutCat);
    attn_kernel<<<(EE * HH + 255) / 256, 256, 0, stream>>>(src, dst, edge_feat, W_edge, OutCat, attnbuf, smaxkey);
    expsum_kernel<<<(EE * HH + 255) / 256, 256, 0, stream>>>(dst, smaxkey, attnbuf, ssum);
    scatter_kernel<<<EE, 256, 0, stream>>>(src, dst, edge_feat, OutCat, W_msg, attnbuf, ssum, accum);
    mean_kernel<<<(NN * OUTF + 255) / 256, 256, 0, stream>>>(accum, out);
}
